// SGC_LSTM_7799660610248
// MI455X (gfx1250) — matrix operand order repaired and checked by reading, not compiled, not run on hardware
//
#include <hip/hip_runtime.h>

// ---------------- problem constants (from reference) ----------------
#define K_G   512
#define K_T   25
#define K_P   10
#define K_M   5
#define K_E   32
#define K_D0  64
#define K_HID 32
#define K_H   128
#define K_NE  40
#define K_N   (K_G * K_T * K_P)   /* 128000 nodes  */
#define K_B   (K_G * K_P)         /* 5120 LSTM batch */

// ---------------- types ----------------
typedef __attribute__((ext_vector_type(16))) __bf16 v16bf_t;
typedef __attribute__((ext_vector_type(8)))  float  v8f_t;
typedef unsigned int u32x4_t __attribute__((ext_vector_type(4)));

union V16U { v16bf_t v; u32x4_t q[2]; };

__device__ __forceinline__ unsigned short f2bf(float f) {
  unsigned int u = __float_as_uint(f);
  u += 0x7FFFu + ((u >> 16) & 1u);           // round-to-nearest-even
  return (unsigned short)(u >> 16);
}

// ---------------------------------------------------------------------------
// WMMA GEMM: C[M x N](f32) = A[M x Kpad](bf16 row-major) * Bp(pre-swizzled bf16)
//            (+bias) (optional fused row l2-normalize, valid for N==32)
// One wave computes a 16x32 tile of C; K consumed in chunks of 32.
// ---------------------------------------------------------------------------
__global__ __launch_bounds__(256)
void gemm_wmma_kernel(const unsigned short* __restrict__ A,
                      const unsigned short* __restrict__ Bp,
                      const float* __restrict__ bias,
                      float* C,
                      int Mrows, int Kpad, int Ncols, int do_l2) {
  const int lane = threadIdx.x & 31;
  const int wave = blockIdx.x * (blockDim.x >> 5) + (threadIdx.x >> 5);
  const int tiles_m  = Mrows >> 4;
  const int tiles_n2 = Ncols >> 5;
  if (wave >= tiles_m * tiles_n2) return;
  const int mt  = wave % tiles_m;
  const int nt0 = (wave / tiles_m) * 2;
  const int Kc  = Kpad >> 5;
  const int hf  = lane >> 4;
  const int l15 = lane & 15;
  const int n0  = nt0 * 16 + l15;
  const int n1  = n0 + 16;

  v8f_t acc0, acc1;
#pragma unroll
  for (int r = 0; r < 8; ++r) { acc0[r] = 0.0f; acc1[r] = 0.0f; }

  const u32x4_t* a4 = (const u32x4_t*)(A + (size_t)(mt * 16 + l15) * Kpad);
  const u32x4_t* b4 = (const u32x4_t*)Bp;
  const size_t bslot0 = ((size_t)nt0 * Kc * 32 + lane) * 2;
  const size_t bslot1 = ((size_t)(nt0 + 1) * Kc * 32 + lane) * 2;

  for (int c = 0; c < Kc; ++c) {
    V16U a, b0, b1;
    a.q[0] = a4[c * 4 + hf];          // K = kk + 8*hf    .. +7
    a.q[1] = a4[c * 4 + 2 + hf];      // K = kk + 16+8*hf .. +7
    const size_t bo = (size_t)c * 64; // 32 lanes * 2 u32x4 per chunk
    b0.q[0] = b4[bslot0 + bo];
    b0.q[1] = b4[bslot0 + bo + 1];
    b1.q[0] = b4[bslot1 + bo];
    b1.q[1] = b4[bslot1 + bo + 1];
    acc0 = __builtin_amdgcn_wmma_f32_16x16x32_bf16(false, a.v, false, b0.v,
                                                   (short)0, acc0, false, false);
    acc1 = __builtin_amdgcn_wmma_f32_16x16x32_bf16(false, a.v, false, b1.v,
                                                   (short)0, acc1, false, false);
  }

  if (bias) {
    const float bb0 = bias[n0];
    const float bb1 = bias[n1];
#pragma unroll
    for (int r = 0; r < 8; ++r) { acc0[r] += bb0; acc1[r] += bb1; }
  }

  if (do_l2) {
#pragma unroll
    for (int r = 0; r < 8; ++r) {
      float ss = acc0[r] * acc0[r] + acc1[r] * acc1[r];
      ss += __shfl_xor(ss, 1, 32);
      ss += __shfl_xor(ss, 2, 32);
      ss += __shfl_xor(ss, 4, 32);
      ss += __shfl_xor(ss, 8, 32);
      const float inv = 1.0f / fmaxf(sqrtf(ss), 1e-12f);
      acc0[r] *= inv;
      acc1[r] *= inv;
    }
  }

#pragma unroll
  for (int r = 0; r < 8; ++r) {
    const size_t row = (size_t)(mt * 16 + r + 8 * hf);
    C[row * Ncols + n0] = acc0[r];
    C[row * Ncols + n1] = acc1[r];
  }
}

// ---------------------------------------------------------------------------
// Pre-swizzle B[K x N] (f32) into WMMA B 32x16 lane layout, bf16, K zero-pad.
// Layout: [ntile][kchunk][lane][16 halfs].
// ---------------------------------------------------------------------------
__global__ void prep_b_kernel(const float* __restrict__ W,
                              unsigned short* __restrict__ Bp,
                              int Kact, int Kpad, int Ncols, int transpose) {
  const int idx = blockIdx.x * blockDim.x + threadIdx.x;
  const int Kc = Kpad >> 5;
  const int total = Ncols * Kpad;
  if (idx >= total) return;
  const int per_tile = Kc * 512;
  const int nt = idx / per_tile;
  int rem = idx - nt * per_tile;
  const int c = rem >> 9;
  rem &= 511;
  const int lane = rem >> 4;
  const int j = rem & 15;
  const int v = j >> 1, odd = j & 1;
  const int k = c * 32 + 8 * (lane >> 4) + (j < 8 ? j : 8 + j);
  const int n = nt * 16 + (lane & 15);
  float val = 0.0f;
  if (k < Kact)
    val = transpose ? W[(size_t)n * Kact + k] : W[(size_t)k * Ncols + n];
  Bp[idx] = f2bf(val);
}

// ---------------------------------------------------------------------------
// Combined LSTM weight swizzle for all T: B[224 x 512] per t where rows 0..95
// are W_ih^T (66 valid, zero-pad) and rows 96..223 are W_hh^T.
// ---------------------------------------------------------------------------
__global__ void prep_lstm_w_kernel(const float* __restrict__ W_ih,
                                   const float* __restrict__ W_hh,
                                   unsigned short* __restrict__ Bp) {
  const int per_t = 224 * 512;
  const long long idx = (long long)blockIdx.x * blockDim.x + threadIdx.x;
  if (idx >= (long long)K_T * per_t) return;
  const int t = (int)(idx / per_t);
  int rem = (int)(idx % per_t);
  const int per_tile = 7 * 512;     // Kc * 512
  const int nt = rem / per_tile;
  rem -= nt * per_tile;
  const int c = rem >> 9;
  rem &= 511;
  const int lane = rem >> 4;
  const int j = rem & 15;
  const int v = j >> 1, odd = j & 1;
  const int k = c * 32 + 8 * (lane >> 4) + (j < 8 ? j : 8 + j);
  const int n = nt * 16 + (lane & 15);
  float val = 0.0f;
  if (k < 96) {
    if (k < 66) val = W_ih[((size_t)t * 512 + n) * 66 + k];
  } else {
    val = W_hh[((size_t)t * 512 + n) * 128 + (k - 96)];
  }
  Bp[idx] = f2bf(val);
}

// ---------------------------------------------------------------------------
// Base SAGE features via LDS float atomics: A rows = [mean(64) | self(64)] bf16.
// ---------------------------------------------------------------------------
__global__ __launch_bounds__(128)
void base_feats_kernel(const float* __restrict__ x0,
                       const int* __restrict__ ps, const int* __restrict__ pd,
                       const int* __restrict__ ns, const int* __restrict__ nd,
                       unsigned short* __restrict__ Apos,
                       unsigned short* __restrict__ Aneg) {
  __shared__ float xs[K_P][K_D0];
  __shared__ float ap[K_P][K_D0];
  __shared__ float an[K_P][K_D0];
  __shared__ float cp[K_P], cn[K_P];
  const int st = blockIdx.x;           // g*T + t
  const int tid = threadIdx.x;
  for (int i = tid; i < K_P * K_D0; i += 128) {
    ((float*)xs)[i] = x0[(size_t)st * K_P * K_D0 + i];
    ((float*)ap)[i] = 0.0f;
    ((float*)an)[i] = 0.0f;
  }
  if (tid < K_P) { cp[tid] = 0.0f; cn[tid] = 0.0f; }
  __syncthreads();
  const int eb = st * K_NE;
  for (int idx = tid; idx < K_NE * K_D0; idx += 128) {
    const int e = idx >> 6, d = idx & 63;
    atomicAdd(&ap[ps[eb + e]][d], xs[pd[eb + e]][d]);  // seg_mean(x[pc], pr)
    atomicAdd(&an[ns[eb + e]][d], xs[nd[eb + e]][d]);
  }
  for (int e = tid; e < K_NE; e += 128) {
    atomicAdd(&cp[ps[eb + e]], 1.0f);
    atomicAdd(&cn[ns[eb + e]], 1.0f);
  }
  __syncthreads();
  for (int i = tid; i < K_P * 2 * K_D0; i += 128) {
    const int p = i >> 7, k = i & 127;
    const size_t row = (size_t)(st * K_P + p) * 128;
    float vp, vn;
    if (k < K_D0) {
      vp = ap[p][k] / fmaxf(cp[p], 1.0f);
      vn = an[p][k] / fmaxf(cn[p], 1.0f);
    } else {
      vp = xs[p][k - K_D0];
      vn = vp;
    }
    Apos[row + k] = f2bf(vp);
    Aneg[row + k] = f2bf(vn);
  }
}

// ---------------------------------------------------------------------------
// Deep-layer feature builder: 6 segment means + concat -> 224-wide bf16 rows.
// ---------------------------------------------------------------------------
__global__ __launch_bounds__(128)
void deep_feats_kernel(const float* __restrict__ hp, const float* __restrict__ hn,
                       const int* __restrict__ ps, const int* __restrict__ pd,
                       const int* __restrict__ ns, const int* __restrict__ nd,
                       unsigned short* __restrict__ Fp,
                       unsigned short* __restrict__ Fn) {
  __shared__ float hps[K_P][K_HID], hns[K_P][K_HID];
  __shared__ float a_ppi[K_P][K_HID], a_ppo[K_P][K_HID];
  __shared__ float a_nni[K_P][K_HID], a_nno[K_P][K_HID];
  __shared__ float a_npi[K_P][K_HID], a_pni[K_P][K_HID];
  __shared__ float cip[K_P], cop[K_P], cin[K_P], con[K_P];
  const int st = blockIdx.x;
  const int tid = threadIdx.x;
  for (int i = tid; i < K_P * K_HID; i += 128) {
    ((float*)hps)[i] = hp[(size_t)st * K_P * K_HID + i];
    ((float*)hns)[i] = hn[(size_t)st * K_P * K_HID + i];
    ((float*)a_ppi)[i] = 0.0f; ((float*)a_ppo)[i] = 0.0f;
    ((float*)a_nni)[i] = 0.0f; ((float*)a_nno)[i] = 0.0f;
    ((float*)a_npi)[i] = 0.0f; ((float*)a_pni)[i] = 0.0f;
  }
  if (tid < K_P) { cip[tid] = cop[tid] = cin[tid] = con[tid] = 0.0f; }
  __syncthreads();
  const int eb = st * K_NE;
  for (int idx = tid; idx < K_NE * K_HID; idx += 128) {
    const int e = idx >> 5, d = idx & 31;
    const int s  = ps[eb + e], t  = pd[eb + e];
    atomicAdd(&a_ppi[s][d], hps[t][d]);   // app_in
    atomicAdd(&a_ppo[t][d], hps[s][d]);   // app_out
    atomicAdd(&a_npi[s][d], hns[t][d]);   // anp_in
    const int s2 = ns[eb + e], t2 = nd[eb + e];
    atomicAdd(&a_nni[s2][d], hns[t2][d]); // ann_in
    atomicAdd(&a_nno[t2][d], hns[s2][d]); // ann_out
    atomicAdd(&a_pni[s2][d], hps[t2][d]); // apn_in
  }
  for (int e = tid; e < K_NE; e += 128) {
    atomicAdd(&cip[ps[eb + e]], 1.0f);
    atomicAdd(&cop[pd[eb + e]], 1.0f);
    atomicAdd(&cin[ns[eb + e]], 1.0f);
    atomicAdd(&con[nd[eb + e]], 1.0f);
  }
  __syncthreads();
  for (int i = tid; i < K_P * 224; i += 128) {
    const int p = i / 224, k = i % 224;
    const int seg = k >> 5, d = k & 31;
    const float icip = 1.0f / fmaxf(cip[p], 1.0f);
    const float icop = 1.0f / fmaxf(cop[p], 1.0f);
    const float icin = 1.0f / fmaxf(cin[p], 1.0f);
    const float icon = 1.0f / fmaxf(con[p], 1.0f);
    float vp, vn;
    switch (seg) {
      case 0: vp = a_ppi[p][d] * icip; vn = a_nni[p][d] * icin; break;
      case 1: vp = a_ppo[p][d] * icop; vn = a_nno[p][d] * icon; break;
      case 2: vp = a_nni[p][d] * icin; vn = a_ppi[p][d] * icip; break;
      case 3: vp = a_nno[p][d] * icon; vn = a_ppo[p][d] * icop; break;
      case 4: vp = a_npi[p][d] * icip; vn = a_pni[p][d] * icin; break;
      case 5: vp = hps[p][d];          vn = hns[p][d];          break;
      default: vp = hns[p][d];         vn = hps[p][d];          break;
    }
    const size_t row = (size_t)(st * K_P + p) * 224;
    Fp[row + k] = f2bf(vp);
    Fn[row + k] = f2bf(vn);
  }
}

// ---------------------------------------------------------------------------
// Build LSTM inputs: X[t][b=g*P+p][96] = [hp(32) | hn(32) | add_info(2) | 0-pad]
// ---------------------------------------------------------------------------
__global__ void build_seq_kernel(const float* __restrict__ hp,
                                 const float* __restrict__ hn,
                                 const float* __restrict__ add_info,
                                 unsigned short* __restrict__ Xall) {
  const long long idx = (long long)blockIdx.x * blockDim.x + threadIdx.x;
  const long long total = (long long)K_T * K_B * 96;
  if (idx >= total) return;
  const int k = (int)(idx % 96);
  const long long row = idx / 96;
  const int t = (int)(row / K_B);
  const int r = (int)(row % K_B);
  const int g = r / K_P, p = r % K_P;
  const int node = (g * K_T + t) * K_P + p;
  float v;
  if (k < 32)       v = hp[(size_t)node * 32 + k];
  else if (k < 64)  v = hn[(size_t)node * 32 + (k - 32)];
  else if (k < 66)  v = add_info[(((size_t)g * K_M + (t / (K_T / K_M))) * K_P + p) * 2 + (k - 64)];
  else              v = 0.0f;
  Xall[idx] = f2bf(v);
}

__global__ void add_bias_kernel(const float* __restrict__ a,
                                const float* __restrict__ b,
                                float* __restrict__ o, int n) {
  const int i = blockIdx.x * blockDim.x + threadIdx.x;
  if (i < n) o[i] = a[i] + b[i];
}

__global__ void f32_to_bf16_kernel(const float* __restrict__ in,
                                   unsigned short* __restrict__ out, int n) {
  const int i = blockIdx.x * blockDim.x + threadIdx.x;
  if (i < n) out[i] = f2bf(in[i]);
}

// ---------------------------------------------------------------------------
// Fused LSTM step: one block owns a 16-row strip of the batch.
//  1) async-DMA the strip's A rows [x_t(96) | h(128)] bf16 into LDS
//     (global_load_async_to_lds_b128 + s_wait_asynccnt, CDNA5 async path)
//  2) 8 waves x 4 WMMA column-tiles -> full 16x512 gates (K = 7 chunks of 32)
//  3) gates through LDS; in-block cell update; h written back in place (bf16)
// ---------------------------------------------------------------------------
__global__ __launch_bounds__(256)
void lstm_step_kernel(const unsigned short* __restrict__ Xt,  // [5120][96]
                      const unsigned short* __restrict__ Bp,  // [32][7][32][16]
                      const float* __restrict__ bias,         // [512]
                      const float* __restrict__ c_in,
                      float* __restrict__ c_out,
                      unsigned short* H) {                    // [5120][128] in/out
  __shared__ __align__(16) unsigned short smemA[16 * 224];    // 448B per row
  __shared__ float smemG[16][512];
  const int strip = blockIdx.x;
  const int tid = threadIdx.x;
  const int lane = tid & 31;
  const int w = tid >> 5;
  const int hf = lane >> 4, l15 = lane & 15;

  // prefetch this wave's weight stream (emits global_prefetch_b8)
  __builtin_prefetch(Bp + (size_t)(4 * w) * 7 * 512, 0, 1);
  __builtin_prefetch(Bp + (size_t)(4 * w + 2) * 7 * 512, 0, 1);

  // ---- stage A rows into LDS with async DMA loads ----
  for (int s = tid; s < 16 * 28; s += 256) {   // 16 rows x (12 X + 16 H) 16B segs
    const int row = s / 28, seg = s % 28;
    const unsigned lds = (unsigned)(size_t)smemA +
        (unsigned)(row * 448 + (seg < 12 ? seg * 16 : 192 + (seg - 12) * 16));
    const void* gsrc = (seg < 12)
        ? (const void*)((const char*)(Xt + (size_t)(strip * 16 + row) * 96) + seg * 16)
        : (const void*)((const char*)(H + (size_t)(strip * 16 + row) * 128) + (seg - 12) * 16);
    asm volatile("global_load_async_to_lds_b128 %0, %1, off"
                 :: "v"(lds), "v"(gsrc) : "memory");
  }
  asm volatile("s_wait_asynccnt 0x0" ::: "memory");
  __syncthreads();

  // ---- gates: wave w computes cols [64w, 64w+64) ----
  v8f_t acc[4];
#pragma unroll
  for (int i = 0; i < 4; ++i)
#pragma unroll
    for (int r = 0; r < 8; ++r) acc[i][r] = 0.0f;

  const char* arow = (const char*)smemA + l15 * 448;
  const u32x4_t* b4 = (const u32x4_t*)Bp;
  for (int c = 0; c < 7; ++c) {
    V16U a;
    a.q[0] = *(const u32x4_t*)(arow + c * 64 + 16 * hf);
    a.q[1] = *(const u32x4_t*)(arow + c * 64 + 32 + 16 * hf);
#pragma unroll
    for (int i = 0; i < 4; ++i) {
      const int nt = 4 * w + i;
      V16U b;
      const size_t slot = ((size_t)(nt * 7 + c) * 32 + lane) * 2;
      b.q[0] = b4[slot];
      b.q[1] = b4[slot + 1];
      acc[i] = __builtin_amdgcn_wmma_f32_16x16x32_bf16(false, a.v, false, b.v,
                                                       (short)0, acc[i], false, false);
    }
  }

  // bias + park gates in LDS (C layout: row = r + 8*hf)
#pragma unroll
  for (int i = 0; i < 4; ++i) {
    const int n = (4 * w + i) * 16 + l15;
    const float bb = bias[n];
#pragma unroll
    for (int r = 0; r < 8; ++r) smemG[r + 8 * hf][n] = acc[i][r] + bb;
  }
  __syncthreads();

  // ---- cell update for this strip; h overwritten in place (safe: staged) ----
  for (int e = tid; e < 16 * 128; e += 256) {
    const int row = e >> 7, j = e & 127;
    const float ig = smemG[row][j];
    const float fg = smemG[row][128 + j];
    const float gg = smemG[row][256 + j];
    const float og = smemG[row][384 + j];
    const float si = 1.0f / (1.0f + expf(-ig));
    const float sf = 1.0f / (1.0f + expf(-fg));
    const float so = 1.0f / (1.0f + expf(-og));
    const size_t gidx = (size_t)(strip * 16 + row) * 128 + j;
    const float c = sf * c_in[gidx] + si * tanhf(gg);
    c_out[gidx] = c;
    H[gidx] = f2bf(so * tanhf(c));
  }
}

// ---------------- host-side helpers ----------------
static inline void launch_gemm(const unsigned short* A, const unsigned short* Bp,
                               const float* bias, float* C,
                               int M, int Kpad, int N, int do_l2, hipStream_t s) {
  const int waves = (M >> 4) * (N >> 5);
  gemm_wmma_kernel<<<(waves + 7) / 8, 256, 0, s>>>(A, Bp, bias, C, M, Kpad, N, do_l2);
}
static inline void launch_prep(const float* W, unsigned short* Bp,
                               int Kact, int Kpad, int N, int tr, hipStream_t s) {
  const int total = N * Kpad;
  prep_b_kernel<<<(total + 255) / 256, 256, 0, s>>>(W, Bp, Kact, Kpad, N, tr);
}

extern "C" void kernel_launch(void* const* d_in, const int* in_sizes, int n_in,
                              void* d_out, int out_size, void* d_ws, size_t ws_size,
                              hipStream_t stream) {
  (void)in_sizes; (void)n_in; (void)out_size; (void)ws_size;
  const float* x0       = (const float*)d_in[0];
  const float* add_info = (const float*)d_in[1];
  const float* Wb_pos   = (const float*)d_in[2];
  const float* bb_pos   = (const float*)d_in[3];
  const float* Wb_neg   = (const float*)d_in[4];
  const float* bb_neg   = (const float*)d_in[5];
  const float* Wd_pos   = (const float*)d_in[6];
  const float* bd_pos   = (const float*)d_in[7];
  const float* Wd_neg   = (const float*)d_in[8];
  const float* bd_neg   = (const float*)d_in[9];
  const float* W_ih     = (const float*)d_in[10];
  const float* W_hh     = (const float*)d_in[11];
  const float* b_ih     = (const float*)d_in[12];
  const float* b_hh     = (const float*)d_in[13];
  const float* hx0      = (const float*)d_in[14];
  const float* cx0      = (const float*)d_in[15];
  const float* Wf       = (const float*)d_in[16];
  const float* bfv      = (const float*)d_in[17];
  const int*   ps       = (const int*)d_in[18];
  const int*   pd       = (const int*)d_in[19];
  const int*   ns       = (const int*)d_in[20];
  const int*   nd       = (const int*)d_in[21];
  float* out = (float*)d_out;

  // ---- workspace layout (256B aligned; peak ~153 MB) ----
  char* ws = (char*)d_ws;
  size_t off = 0;
  auto alloc = [&](size_t b) { size_t o = off; off += (b + 255) & ~(size_t)255; return o; };
  float* hp = (float*)(ws + alloc((size_t)K_N * 32 * 4));
  float* hn = (float*)(ws + alloc((size_t)K_N * 32 * 4));
  char* big = ws + alloc(114688000);  // phase-shared scratch
  // phase 1: base SAGE A matrices
  unsigned short* Apos = (unsigned short*)big;
  unsigned short* Aneg = (unsigned short*)(big + (size_t)K_N * 128 * 2);
  // phase 2: deep-layer features
  unsigned short* Fp = (unsigned short*)big;
  unsigned short* Fn = (unsigned short*)(big + (size_t)K_N * 224 * 2);
  // phase 3: LSTM
  unsigned short* Xall = (unsigned short*)big;                  // 24,576,000 B
  const size_t xbytes = (size_t)K_T * K_B * 96 * 2;
  unsigned short* Hbf = (unsigned short*)(big + xbytes);        // 1,310,720 B
  const size_t hbytes = (size_t)K_B * K_H * 2;
  float* cbuf = (float*)(big + xbytes + hbytes);                // 2,621,440 B
  // small pre-swizzled weights
  unsigned short* WbPp = (unsigned short*)(ws + alloc(128 * 32 * 2));
  unsigned short* WbPn = (unsigned short*)(ws + alloc(128 * 32 * 2));
  unsigned short* WdPp = (unsigned short*)(ws + alloc(224 * 32 * 2));
  unsigned short* WdPn = (unsigned short*)(ws + alloc(224 * 32 * 2));
  unsigned short* WfP  = (unsigned short*)(ws + alloc(128 * 32 * 2));
  unsigned short* WcP  = (unsigned short*)(ws + alloc((size_t)K_T * 224 * 512 * 2));
  float* biasC = (float*)(ws + alloc((size_t)K_T * 512 * 4));

  // ---- base signed SAGE ----
  base_feats_kernel<<<K_G * K_T, 128, 0, stream>>>(x0, ps, pd, ns, nd, Apos, Aneg);
  launch_prep(Wb_pos, WbPp, 128, 128, 32, 0, stream);
  launch_prep(Wb_neg, WbPn, 128, 128, 32, 0, stream);
  launch_gemm(Apos, WbPp, bb_pos, hp, K_N, 128, 32, 1, stream);
  launch_gemm(Aneg, WbPn, bb_neg, hn, K_N, 128, 32, 1, stream);

  // ---- two deep signed layers ----
  for (int l = 0; l < 2; ++l) {
    deep_feats_kernel<<<K_G * K_T, 128, 0, stream>>>(hp, hn, ps, pd, ns, nd, Fp, Fn);
    launch_prep(Wd_pos + (size_t)l * 224 * 32, WdPp, 224, 224, 32, 0, stream);
    launch_prep(Wd_neg + (size_t)l * 224 * 32, WdPn, 224, 224, 32, 0, stream);
    launch_gemm(Fp, WdPp, bd_pos + l * 32, hp, K_N, 224, 32, 1, stream);
    launch_gemm(Fn, WdPn, bd_neg + l * 32, hn, K_N, 224, 32, 1, stream);
  }

  // ---- LSTM inputs + combined per-timestep weight swizzle ----
  {
    const long long total = (long long)K_T * K_B * 96;
    build_seq_kernel<<<(unsigned)((total + 255) / 256), 256, 0, stream>>>(hp, hn, add_info, Xall);
  }
  {
    const long long total = (long long)K_T * 224 * 512;
    prep_lstm_w_kernel<<<(unsigned)((total + 255) / 256), 256, 0, stream>>>(W_ih, W_hh, WcP);
  }
  add_bias_kernel<<<(K_T * 512 + 255) / 256, 256, 0, stream>>>(b_ih, b_hh, biasC, K_T * 512);
  f32_to_bf16_kernel<<<(K_B * K_H + 255) / 256, 256, 0, stream>>>(hx0, Hbf, K_B * K_H);

  // ---- recurrence: 25 fused step kernels (single launch per timestep) ----
  for (int t = 0; t < K_T; ++t) {
    lstm_step_kernel<<<K_B / 16, 256, 0, stream>>>(
        Xall + (size_t)t * K_B * 96,
        WcP + (size_t)t * 224 * 512,
        biasC + (size_t)t * 512,
        (t == 0) ? cx0 : cbuf, cbuf, Hbf);
  }

  // ---- final projection h_last @ Wf + bf -> [G, P, E] ----
  launch_prep(Wf, WfP, 128, 128, 32, 0, stream);
  launch_gemm(Hbf, WfP, bfv, out, K_B, 128, 32, 0, stream);
}
